// FlowmatchingActionHead_50190987821416
// MI455X (gfx1250) — matrix operand order repaired and checked by reading, not compiled, not run on hardware
//
#include <hip/hip_runtime.h>

typedef __attribute__((ext_vector_type(16))) _Float16 v16h;
typedef __attribute__((ext_vector_type(8)))  _Float16 v8h;
typedef __attribute__((ext_vector_type(2)))  _Float16 v2h;
typedef __attribute__((ext_vector_type(8)))  float    v8f;
typedef __attribute__((ext_vector_type(4)))  float    v4f;

#define NB 128   // batch
#define NT 256   // tokens
#define ND 1536  // in dim
#define NH 1024  // hidden
#define NO 16    // out dim

// LDS layout (aliased; phases separated by barriers):
//   [0      .. 20480)  Alds[2][128][40]   x tile, double-buffered      (K-loop)
//   [20480  .. 40960)  Blds[2][128][40]   W1 tile transposed, 2 bufs   (K-loop)
//   [0      .. 34816)  Hlds[128][136]     relu(h) tile                 (epilogue)
//   [40960  .. 45056)  W2t[16][128]       W2 slice transposed          (epilogue)
#define A_OFF   0
#define B_OFF   20480
#define W2_OFF  40960
#define SMEM_SZ 45056

// Split workgroup barrier (CDNA5 S_BARRIER_SIGNAL/WAIT on barrier -1).
// Arrive: drain this wave's LDS ops (stores visible to consumers, and our own
// reads of the soon-to-be-overwritten buffer retired), then signal.
__device__ __forceinline__ void wg_barrier_arrive() {
    asm volatile("s_wait_dscnt 0x0\n\t"
                 "s_barrier_signal -1" ::: "memory");
}
__device__ __forceinline__ void wg_barrier_wait() {
    asm volatile("s_barrier_wait -1" ::: "memory");
}

__global__ __launch_bounds__(256, 1)
void catmlp_fused(const float* __restrict__ x,  const float* __restrict__ W1,
                  const float* __restrict__ b1, const float* __restrict__ W2,
                  const float* __restrict__ b2, const int* __restrict__ cat_ids,
                  float* __restrict__ out)
{
    __shared__ __align__(16) unsigned char smem_raw[SMEM_SZ];
    auto* Alds = reinterpret_cast<_Float16 (*)[128][40]>(smem_raw + A_OFF);
    auto* Blds = reinterpret_cast<_Float16 (*)[128][40]>(smem_raw + B_OFF);
    auto* Hlds = reinterpret_cast<_Float16 (*)[136]>(smem_raw);
    auto* W2t  = reinterpret_cast<_Float16 (*)[128]>(smem_raw + W2_OFF);

    const int tid   = threadIdx.x;
    const int lane  = tid & 31;
    const int wave  = tid >> 5;        // 0..7
    const int l16   = lane & 15;
    const int lhalf = lane >> 4;       // 0 or 1

    const int b  = blockIdx.x >> 1;
    const int t0 = (blockIdx.x & 1) * 128;
    const int c  = cat_ids[b];

    // layer1 wave tiling: 4x2 waves -> 32-row x 64-col patch per wave
    const int wm = wave >> 1;
    const int wn = wave & 1;
    const int R  = wm * 32;
    const int Cc = wn * 64;

    // staging roles
    const int arow  = tid >> 1;        // 0..127  (x tile row)
    const int akoff = (tid & 1) * 16;  // 0 or 16 (x tile k half)
    const int kp    = tid >> 4;        // 0..15   (W1 k-pair)
    const int hg    = tid & 15;        // 0..15   (h-group of 8)

    const float* xrow   = x  + ((size_t)b * NT + t0 + arow) * ND;
    const float* w1base = W1 + (size_t)c * ND * NH;

    v8f out_acc = {};   // 16 rows x 16 cols of final output per wave

    for (int hs = 0; hs < NH / 128; ++hs) {
        const int h0 = hs * 128;

        v8f acc[2][4];
        #pragma unroll
        for (int i = 0; i < 2; ++i)
            #pragma unroll
            for (int j = 0; j < 4; ++j) acc[i][j] = (v8f){};

        // ---- hoisted epilogue loads: overlap the whole K loop ----
        float biasv[4];
        #pragma unroll
        for (int j = 0; j < 4; ++j)
            biasv[j] = b1[c * NH + h0 + Cc + 16 * j + l16];
        {   // W2t[o][h_local], strided gather (o fastest dim in W2)
            const int o = tid >> 4;
            const float* p = W2 + ((size_t)c * NH + h0 + hg * 8) * NO + o;
            #pragma unroll
            for (int j = 0; j < 8; ++j)
                W2t[o][hg * 8 + j] = (_Float16)p[(size_t)j * NO];
        }

        // ---- double-buffered staging: regs for next tile, LDS for current ----
        v4f ax0, ax1, ax2, ax3;        // x tile fragment (16 f32)
        v4f bw0, bw1, bw2, bw3;        // W1 tile fragment (2 rows x 8 cols)

        auto load_tile = [&](int kb) {
            const float* pA = xrow + kb + akoff;
            ax0 = *(const v4f*)(pA + 0);
            ax1 = *(const v4f*)(pA + 4);
            ax2 = *(const v4f*)(pA + 8);
            ax3 = *(const v4f*)(pA + 12);
            const float* pB = w1base + (size_t)(kb + 2 * kp) * NH + h0 + hg * 8;
            bw0 = *(const v4f*)(pB);
            bw1 = *(const v4f*)(pB + 4);
            bw2 = *(const v4f*)(pB + NH);
            bw3 = *(const v4f*)(pB + NH + 4);
        };
        auto store_tile = [&](int buf) {
            v8h lo, hi;
            #pragma unroll
            for (int e = 0; e < 4; ++e) {
                lo[e]     = (_Float16)ax0[e];
                lo[e + 4] = (_Float16)ax1[e];
                hi[e]     = (_Float16)ax2[e];
                hi[e + 4] = (_Float16)ax3[e];
            }
            *(v8h*)&Alds[buf][arow][akoff]     = lo;
            *(v8h*)&Alds[buf][arow][akoff + 8] = hi;
            #pragma unroll
            for (int j = 0; j < 8; ++j) {
                float va = (j < 4) ? bw0[j] : bw1[j - 4];
                float vb = (j < 4) ? bw2[j] : bw3[j - 4];
                v2h pk = {(_Float16)va, (_Float16)vb};
                *(v2h*)&Blds[buf][hg * 8 + j][2 * kp] = pk;
            }
        };

        load_tile(0);
        store_tile(0);
        wg_barrier_arrive();

        // -------- layer1 K loop: split barrier, one round per step --------
        for (int kb = 0, step = 0; kb < ND; kb += 32, ++step) {
            const int  cur       = step & 1;
            const bool have_next = (kb + 32) < ND;

            if (have_next) load_tile(kb + 32);   // issue while others arrive
            if (kb + 96 < ND) {                  // warm L2 ~3 steps out
                __builtin_prefetch(xrow + kb + 96 + akoff, 0, 3);
                __builtin_prefetch(w1base + (size_t)(kb + 96 + 2 * kp) * NH
                                   + h0 + hg * 8, 0, 3);
            }

            wg_barrier_wait();                   // buffer `cur` now valid

            // B fragments (shared across both row tiles of this wave)
            v16h bfrag[4];
            {
                const int kb1 = lhalf * 8;   // K {0-7,16-23} / {8-15,24-31}
                #pragma unroll
                for (int j = 0; j < 4; ++j) {
                    const int n = Cc + 16 * j + l16;
                    v8h blo = *(const v8h*)&Blds[cur][n][kb1];
                    v8h bhi = *(const v8h*)&Blds[cur][n][kb1 + 16];
                    bfrag[j] = __builtin_shufflevector(blo, bhi,
                        0,1,2,3,4,5,6,7,8,9,10,11,12,13,14,15);
                }
            }
            #pragma unroll
            for (int i = 0; i < 2; ++i) {
                const int lrow = R + 16 * i + l16;
                const int ka   = lhalf * 8;  // K {0-7,16-23} / {8-15,24-31}
                v8h alo = *(const v8h*)&Alds[cur][lrow][ka];
                v8h ahi = *(const v8h*)&Alds[cur][lrow][ka + 16];
                v16h afrag = __builtin_shufflevector(alo, ahi,
                    0,1,2,3,4,5,6,7,8,9,10,11,12,13,14,15);
                #pragma unroll
                for (int j = 0; j < 4; ++j) {
                    acc[i][j] = __builtin_amdgcn_wmma_f32_16x16x32_f16(
                        false, afrag, false, bfrag[j],
                        (short)0, acc[i][j], false, false);
                }
            }

            if (have_next) store_tile(1 - cur);  // fill alternate buffer
            wg_barrier_arrive();                 // stores drained, signal
        }
        wg_barrier_wait();  // all waves done with staging buffers (aliased)

        // ---------------- bias + relu -> Hlds (aliases staging LDS) ----------
        #pragma unroll
        for (int i = 0; i < 2; ++i) {
            #pragma unroll
            for (int j = 0; j < 4; ++j) {
                const int ncol = Cc + 16 * j + l16;
                #pragma unroll
                for (int v = 0; v < 8; ++v) {
                    float val = acc[i][j][v] + biasv[j];
                    val = val > 0.f ? val : 0.f;
                    Hlds[R + 16 * i + v + 8 * lhalf][ncol] = (_Float16)val;
                }
            }
        }
        __syncthreads();

        // ---------------- layer2: out_acc += relu(h) * W2_slice --------------
        // wave w owns output rows 16w..16w+15; K = 128 h-columns, 4 WMMA steps
        {
            const int lrow = 16 * wave + l16;
            #pragma unroll
            for (int kk = 0; kk < 128; kk += 32) {
                const int ka = kk + lhalf * 8;
                v8h alo = *(const v8h*)&Hlds[lrow][ka];
                v8h ahi = *(const v8h*)&Hlds[lrow][ka + 16];
                v16h afrag = __builtin_shufflevector(alo, ahi,
                    0,1,2,3,4,5,6,7,8,9,10,11,12,13,14,15);
                const int kb2 = kk + lhalf * 8;
                v8h blo = *(const v8h*)&W2t[l16][kb2];
                v8h bhi = *(const v8h*)&W2t[l16][kb2 + 16];
                v16h bfrag = __builtin_shufflevector(blo, bhi,
                    0,1,2,3,4,5,6,7,8,9,10,11,12,13,14,15);
                out_acc = __builtin_amdgcn_wmma_f32_16x16x32_f16(
                    false, afrag, false, bfrag, (short)0, out_acc, false, false);
            }
        }
        __syncthreads();  // protects aliased region before next hs staging
    }

    // ---------------- store: out = out_acc + b2 ----------------
    {
        const int o = l16;
        const float bias2 = b2[c * NO + o];
        #pragma unroll
        for (int v = 0; v < 8; ++v) {
            const int row = 16 * wave + v + 8 * lhalf;
            out[((size_t)b * NT + t0 + row) * NO + o] = out_acc[v] + bias2;
        }
    }
}

extern "C" void kernel_launch(void* const* d_in, const int* in_sizes, int n_in,
                              void* d_out, int out_size, void* d_ws, size_t ws_size,
                              hipStream_t stream) {
    const float* x   = (const float*)d_in[0];
    const float* W1  = (const float*)d_in[1];
    const float* b1  = (const float*)d_in[2];
    const float* W2  = (const float*)d_in[3];
    const float* b2  = (const float*)d_in[4];
    const int*   cid = (const int*)d_in[5];
    float* out = (float*)d_out;

    // 128 samples x 2 token tiles of 128 rows
    catmlp_fused<<<dim3(NB * 2), dim3(256), 0, stream>>>(x, W1, b1, W2, b2, cid, out);
}
